// WPL_routed_all_fc_76321568850311
// MI455X (gfx1250) — matrix operand order repaired and checked by reading, not compiled, not run on hardware
//
#include <hip/hip_runtime.h>

typedef __attribute__((ext_vector_type(16))) _Float16 v16h;
typedef __attribute__((ext_vector_type(8)))  _Float16 v8h;
typedef __attribute__((ext_vector_type(8)))  float    v8f;
typedef __attribute__((ext_vector_type(4)))  float    v4f;
typedef int v4i __attribute__((vector_size(16)));

#define NB 8192

#if defined(__gfx1250__) && __has_builtin(__builtin_amdgcn_global_load_async_to_lds_b128) && __has_builtin(__builtin_amdgcn_s_wait_asynccnt)
#define HAVE_ASYNC 1
#else
#define HAVE_ASYNC 0
#endif

#define GLP  __attribute__((address_space(1)))
#define LDSP __attribute__((address_space(3)))

#if HAVE_ASYNC
// 16-byte async DMA: global -> LDS (tracked by ASYNCcnt)
__device__ __forceinline__ void async_ld16(void* lds_dst, const void* gsrc) {
  const GLP v4i* s = (const GLP v4i*)gsrc;
  LDSP v4i* d = (LDSP v4i*)lds_dst;
  __builtin_amdgcn_global_load_async_to_lds_b128((GLP v4i*)s, d, 0, 0);
}
#define ASYNC_WAIT() __builtin_amdgcn_s_wait_asynccnt(0)
#else
#define ASYNC_WAIT()
#endif

// 16B copy global->LDS (async when available)
__device__ __forceinline__ void cp16(void* lds_dst, const void* gsrc) {
#if HAVE_ASYNC
  async_ld16(lds_dst, gsrc);
#else
  *(v8h*)lds_dst = *(const v8h*)gsrc;
#endif
}

__device__ __forceinline__ v16h make_frag(v8h lo, v8h hi) {
  v16h a;
#pragma unroll
  for (int i = 0; i < 8; ++i) { a[i] = lo[i]; a[i + 8] = hi[i]; }
  return a;
}

__device__ __forceinline__ v8h relu_pack(v8f acc) {
  v8h o;
#pragma unroll
  for (int i = 0; i < 8; ++i) { float v = acc[i]; o[i] = (_Float16)(v > 0.f ? v : 0.f); }
  return o;
}

// ---------------------------------------------------------------------------
// k_prep: one-shot f32 -> f16 weight conversion into workspace.
// ---------------------------------------------------------------------------
__global__ __launch_bounds__(256) void k_prep(
    const float* __restrict__ W1, const float* __restrict__ W2,
    const float* __restrict__ W3, const float* __restrict__ Wc1,
    const float* __restrict__ Wc2, const float* __restrict__ Wc3,
    _Float16* __restrict__ fw1, _Float16* __restrict__ fw2,
    _Float16* __restrict__ fw3, _Float16* __restrict__ cw1,
    _Float16* __restrict__ cw2, _Float16* __restrict__ cw3)
{
  int i = blockIdx.x * 256 + threadIdx.x;
  if (i < 524288) { fw1[i] = (_Float16)W1[i]; return; }
  i -= 524288;
  if (i < 65536) { fw2[i] = (_Float16)W2[i]; return; }
  i -= 65536;
  if (i < 16384) {
    int ex = i >> 10, oc = (i >> 6) & 15, k = i & 63;
    fw3[i] = (oc < 10) ? (_Float16)W3[(ex * 10 + oc) * 64 + k] : (_Float16)0.f;
    return;
  }
  i -= 16384;
  if (i < 1024) {
    int oc = i >> 5, k = i & 31;
    cw1[i] = (k < 27) ? (_Float16)Wc1[oc * 27 + (k % 3) * 9 + (k / 3)] : (_Float16)0.f;
    return;
  }
  i -= 1024;
  if (i < 9216) {
    int oc = i / 288, k = i % 288;
    cw2[i] = (_Float16)Wc2[oc * 288 + (k & 31) * 9 + (k >> 5)];
    return;
  }
  i -= 9216;
  {
    int oc = i / 288, k = i % 288;
    cw3[i] = (_Float16)Wc3[oc * 288 + (k & 31) * 9 + (k >> 5)];
  }
}

// ---------------------------------------------------------------------------
// conv1: 3->32 ch, 32x32, pad1, relu, maxpool2 -> act1 [B][16][16][32] f16 NHWC
// Coalesced input staging (gx fastest); A-fragment shared across oc-tiles.
// ---------------------------------------------------------------------------
__global__ __launch_bounds__(256) void k_conv1(
    const float* __restrict__ x, const _Float16* __restrict__ cw1,
    const float* __restrict__ bias, _Float16* __restrict__ act1)
{
  const int hf  = blockIdx.x & 1;
  const int img = blockIdx.x >> 1;
  const int tid = threadIdx.x;

  __shared__ alignas(16) _Float16 sIn[19][34][4];   // [j][xx][ic], halo+zero pad
  __shared__ alignas(16) _Float16 sWt[32][32];
  __shared__ float sB[32];
  __shared__ alignas(16) _Float16 sOut[32][16][32]; // [oc][y][x]

  {
    const v8h z8 = {};
    for (int i = tid; i < 323; i += 256)            // zero full halo tile
      *(v8h*)(((_Float16*)sIn) + 8 * i) = z8;
  }
  for (int i = tid; i < 64; i += 256)               // weights: 64 x 16B DMA
    cp16(((_Float16*)sWt) + 8 * i, cw1 + 8 * i);
  if (tid < 32) sB[tid] = bias[tid];
  __syncthreads();
  // fill valid cells; gx is the fastest index -> coalesced global reads
  for (int n = tid; n < 18 * 96; n += 256) {
    int gx = n & 31, ic = (n >> 5) % 3, j = n / 96;
    int gy = 16 * hf - 1 + j;
    if (gy >= 0 && gy < 32)
      sIn[j][gx + 1][ic] = (_Float16)x[((img * 3 + ic) * 32 + gy) * 32 + gx];
  }
  ASYNC_WAIT();
  __syncthreads();

  const int wave = tid >> 5, lane = tid & 31;
  const int r = lane & 15, sel = lane >> 4;
  const int oc0 = lane & 15, oc1 = 16 + (lane & 15);
#pragma unroll 2
  for (int tt = wave; tt < 32; tt += 8) {           // 32 pixel tiles
    int yl = tt >> 1;
    int x0 = (tt & 1) << 4;
    v16h a;
#pragma unroll
    for (int h = 0; h < 16; ++h) {
      int k = 16 * sel + h;
      int ic = k % 3, tap = k / 3, dy = tap / 3, dx = tap % 3;
      a[h] = sIn[yl + dy][x0 + r + dx][ic];
    }
    int k0 = sel * 16;
    v16h b0 = make_frag(*(const v8h*)&sWt[oc0][k0], *(const v8h*)&sWt[oc0][k0 + 8]);
    v16h b1 = make_frag(*(const v8h*)&sWt[oc1][k0], *(const v8h*)&sWt[oc1][k0 + 8]);
    v8f acc0, acc1;
#pragma unroll
    for (int i = 0; i < 8; ++i) { acc0[i] = sB[oc0]; acc1[i] = sB[oc1]; }
    acc0 = __builtin_amdgcn_wmma_f32_16x16x32_f16(false, a, false, b0,
                                                  (short)0, acc0, false, false);
    acc1 = __builtin_amdgcn_wmma_f32_16x16x32_f16(false, a, false, b1,
                                                  (short)0, acc1, false, false);
    *(v8h*)&sOut[oc0][yl][x0 + 8 * sel] = relu_pack(acc0);
    *(v8h*)&sOut[oc1][yl][x0 + 8 * sel] = relu_pack(acc1);
  }
  __syncthreads();
  for (int i = tid; i < 8 * 16 * 32; i += 256) {
    int oc = i & 31, px = (i >> 5) & 15, pyl = i >> 9;
    float a0 = (float)sOut[oc][2 * pyl][2 * px];
    float a1 = (float)sOut[oc][2 * pyl][2 * px + 1];
    float a2 = (float)sOut[oc][2 * pyl + 1][2 * px];
    float a3 = (float)sOut[oc][2 * pyl + 1][2 * px + 1];
    float m = fmaxf(fmaxf(a0, a1), fmaxf(a2, a3));
    int py = 8 * hf + pyl;
    act1[((img * 16 + py) * 16 + px) * 32 + oc] = (_Float16)m;
  }
}

// ---------------------------------------------------------------------------
// conv2: 32->32 ch, 16x16 -> pool -> act2 [B][8][8][32] f16. K=288.
// ---------------------------------------------------------------------------
__global__ __launch_bounds__(256) void k_conv2(
    const _Float16* __restrict__ act1, const _Float16* __restrict__ cw2,
    const float* __restrict__ bias, _Float16* __restrict__ act2)
{
  const int img = blockIdx.x, tid = threadIdx.x;
  __shared__ alignas(16) _Float16 sIn[18][18][32];  // halo-padded, [j][i][ic]
  __shared__ alignas(16) _Float16 sWt[32][288];     // [oc][tap*32+ic]
  __shared__ float sB[32];
  __shared__ alignas(16) _Float16 sOut[32][16][16]; // [oc][y][x]

  {
    const v8h z8 = {};
    for (int i = tid; i < 272; i += 256) {
      int c = i >> 2, q = i & 3;
      int j, ii;
      if (c < 18)      { j = 0;      ii = c; }
      else if (c < 36) { j = 17;     ii = c - 18; }
      else if (c < 52) { j = c - 35; ii = 0; }
      else             { j = c - 51; ii = 17; }
      *(v8h*)&sIn[j][ii][8 * q] = z8;
    }
  }
  for (int i = tid; i < 1024; i += 256) {           // interior: 16 rows x 1024B
    int j = 1 + (i >> 6), o = i & 63;
    cp16(&sIn[j][1][0] + 8 * o, act1 + (size_t)img * 8192 + (j - 1) * 512 + 8 * o);
  }
  for (int i = tid; i < 576; i += 256)              // weights: 576 x 16B DMA (4608 halfs)
    cp16(((_Float16*)sWt) + 8 * i, cw2 + 8 * i);
  if (tid < 32) sB[tid] = bias[tid];
  ASYNC_WAIT();
  __syncthreads();

  const int wave = tid >> 5, lane = tid & 31;
  const int r = lane & 15, sel = lane >> 4;
  const int oc0 = lane & 15, oc1 = 16 + (lane & 15);
#pragma unroll
  for (int tt = wave; tt < 16; tt += 8) {           // t = conv row y, x = r
    v8f acc0, acc1;
#pragma unroll
    for (int i = 0; i < 8; ++i) { acc0[i] = sB[oc0]; acc1[i] = sB[oc1]; }
#pragma unroll
    for (int tap = 0; tap < 9; ++tap) {
      int dy = tap / 3, dx = tap % 3;
      int j = tt + dy, ii = r + dx;
      v16h a = make_frag(*(const v8h*)&sIn[j][ii][16 * sel],
                         *(const v8h*)&sIn[j][ii][16 * sel + 8]);
      int k0 = tap * 32 + sel * 16;
      v16h b0 = make_frag(*(const v8h*)&sWt[oc0][k0], *(const v8h*)&sWt[oc0][k0 + 8]);
      v16h b1 = make_frag(*(const v8h*)&sWt[oc1][k0], *(const v8h*)&sWt[oc1][k0 + 8]);
      acc0 = __builtin_amdgcn_wmma_f32_16x16x32_f16(false, a, false, b0,
                                                    (short)0, acc0, false, false);
      acc1 = __builtin_amdgcn_wmma_f32_16x16x32_f16(false, a, false, b1,
                                                    (short)0, acc1, false, false);
    }
    *(v8h*)&sOut[oc0][tt][8 * sel] = relu_pack(acc0);
    *(v8h*)&sOut[oc1][tt][8 * sel] = relu_pack(acc1);
  }
  __syncthreads();
  for (int i = tid; i < 8 * 8 * 32; i += 256) {
    int oc = i & 31, px = (i >> 5) & 7, py = i >> 8;
    float a0 = (float)sOut[oc][2 * py][2 * px];
    float a1 = (float)sOut[oc][2 * py][2 * px + 1];
    float a2 = (float)sOut[oc][2 * py + 1][2 * px];
    float a3 = (float)sOut[oc][2 * py + 1][2 * px + 1];
    act2[((img * 8 + py) * 8 + px) * 32 + oc] =
        (_Float16)fmaxf(fmaxf(a0, a1), fmaxf(a2, a3));
  }
}

// ---------------------------------------------------------------------------
// conv3: 32->32 ch, 8x8 -> pool -> act3 [B][4][4][32] f32 (pre-batchnorm)
// ---------------------------------------------------------------------------
__global__ __launch_bounds__(256) void k_conv3(
    const _Float16* __restrict__ act2, const _Float16* __restrict__ cw3,
    const float* __restrict__ bias, float* __restrict__ act3)
{
  const int img = blockIdx.x, tid = threadIdx.x;
  __shared__ alignas(16) _Float16 sIn[10][10][32];  // halo-padded
  __shared__ alignas(16) _Float16 sWt[32][288];
  __shared__ float sB[32];
  __shared__ alignas(16) _Float16 sOut[32][64];     // [oc][p], p = y*8+x

  {
    const v8h z8 = {};
    for (int i = tid; i < 144; i += 256) {
      int c = i >> 2, q = i & 3;
      int j, ii;
      if (c < 10)      { j = 0;      ii = c; }
      else if (c < 20) { j = 9;      ii = c - 10; }
      else if (c < 28) { j = c - 19; ii = 0; }
      else             { j = c - 27; ii = 9; }
      *(v8h*)&sIn[j][ii][8 * q] = z8;
    }
  }
  for (int i = tid; i < 256; i += 256) {
    int j = 1 + (i >> 5), o = i & 31;
    cp16(&sIn[j][1][0] + 8 * o, act2 + (size_t)img * 2048 + (j - 1) * 256 + 8 * o);
  }
  for (int i = tid; i < 576; i += 256)
    cp16(((_Float16*)sWt) + 8 * i, cw3 + 8 * i);
  if (tid < 32) sB[tid] = bias[tid];
  ASYNC_WAIT();
  __syncthreads();

  const int wave = tid >> 5, lane = tid & 31;
  const int r = lane & 15, sel = lane >> 4;
  {
    int t = wave >> 1, octile = wave & 1;     // 4 px-tiles x 2 oc-tiles
    int p = 16 * t + r, y = p >> 3, xx = p & 7;
    int oc = octile * 16 + (lane & 15);
    v8f acc;
#pragma unroll
    for (int i = 0; i < 8; ++i) acc[i] = sB[oc];
#pragma unroll
    for (int tap = 0; tap < 9; ++tap) {
      int dy = tap / 3, dx = tap % 3;
      int j = y + dy, ii = xx + dx;
      v16h a = make_frag(*(const v8h*)&sIn[j][ii][16 * sel],
                         *(const v8h*)&sIn[j][ii][16 * sel + 8]);
      int k0 = tap * 32 + sel * 16;
      v16h bf = make_frag(*(const v8h*)&sWt[oc][k0], *(const v8h*)&sWt[oc][k0 + 8]);
      acc = __builtin_amdgcn_wmma_f32_16x16x32_f16(false, a, false, bf,
                                                   (short)0, acc, false, false);
    }
    *(v8h*)&sOut[oc][16 * t + 8 * sel] = relu_pack(acc);
  }
  __syncthreads();
  for (int i = tid; i < 4 * 4 * 32; i += 256) {
    int oc = i & 31, px = (i >> 5) & 3, py = i >> 7;
    float a0 = (float)sOut[oc][(2 * py) * 8 + 2 * px];
    float a1 = (float)sOut[oc][(2 * py) * 8 + 2 * px + 1];
    float a2 = (float)sOut[oc][(2 * py + 1) * 8 + 2 * px];
    float a3 = (float)sOut[oc][(2 * py + 1) * 8 + 2 * px + 1];
    act3[((img * 4 + py) * 4 + px) * 32 + oc] = fmaxf(fmaxf(a0, a1), fmaxf(a2, a3));
  }
}

// ---------------------------------------------------------------------------
// BN stats, pass 1: 64 blocks, coalesced v4f reads, deterministic.
// Thread t accumulates channels ((4t)&31)..+3; fixed-order LDS reduction.
// part[blk*64 + c] = sum, part[blk*64 + 32 + c] = sumsq
// ---------------------------------------------------------------------------
__global__ __launch_bounds__(256) void k_bnpart(
    const float* __restrict__ act3, float* __restrict__ part)
{
  const int blk = blockIdx.x, tid = threadIdx.x;
  const float* base = act3 + (size_t)blk * 65536;
  float s0 = 0.f, s1 = 0.f, s2 = 0.f, s3 = 0.f;
  float q0 = 0.f, q1 = 0.f, q2 = 0.f, q3 = 0.f;
  for (int it = 0; it < 64; ++it) {
    v4f v = *(const v4f*)(base + 4 * tid + 1024 * it);
    s0 += v[0]; q0 += v[0] * v[0];
    s1 += v[1]; q1 += v[1] * v[1];
    s2 += v[2]; q2 += v[2] * v[2];
    s3 += v[3]; q3 += v[3] * v[3];
  }
  __shared__ float sS[256][4], sQ[256][4];
  sS[tid][0] = s0; sS[tid][1] = s1; sS[tid][2] = s2; sS[tid][3] = s3;
  sQ[tid][0] = q0; sQ[tid][1] = q1; sQ[tid][2] = q2; sQ[tid][3] = q3;
  __syncthreads();
  if (tid < 32) {
    float ss = 0.f, qq = 0.f;
    for (int m = 0; m < 32; ++m) {
      int t = (tid >> 2) + 8 * m;      // threads with channel group tid>>2
      ss += sS[t][tid & 3];
      qq += sQ[t][tid & 3];
    }
    part[blk * 64 + tid] = ss;
    part[blk * 64 + 32 + tid] = qq;
  }
}

// BN stats, pass 2: fold 64 partials per channel (fixed order), emit scale/shift
__global__ __launch_bounds__(32) void k_bnfinal(
    const float* __restrict__ part, const float* __restrict__ gamma,
    const float* __restrict__ beta, float* __restrict__ stats)
{
  int c = threadIdx.x;
  float s = 0.f, q = 0.f;
  for (int b = 0; b < 64; ++b) {
    s += part[b * 64 + c];
    q += part[b * 64 + 32 + c];
  }
  float n = (float)(NB * 16);
  float mean = s / n;
  float var  = q / n - mean * mean;
  float sc = gamma[c] * rsqrtf(var + 1e-5f);
  stats[c] = sc;
  stats[32 + c] = beta[c] - mean * sc;
}

// ---------------------------------------------------------------------------
// feat: BN apply + NHWC -> reference NCHW-flatten transpose, tiled via LDS.
// 16 samples per block: coalesced v4f loads, contiguous v8h stores.
// ---------------------------------------------------------------------------
__global__ __launch_bounds__(256) void k_feat(
    const float* __restrict__ act3, const float* __restrict__ stats,
    _Float16* __restrict__ feat)
{
  const int b0 = blockIdx.x * 16;
  const int tid = threadIdx.x;
  __shared__ alignas(16) _Float16 sF[16][512];
  for (int i = tid; i < 2048; i += 256) {            // 8192 floats, v4f chunks
    int off = 4 * i;
    v4f v = *(const v4f*)(act3 + (size_t)b0 * 512 + off);
#pragma unroll
    for (int j = 0; j < 4; ++j) {
      int e = off + j;
      int smp = e >> 9, rem = e & 511;
      int c = rem & 31, hw = rem >> 5;               // hw = h*4+w
      sF[smp][c * 16 + hw] = (_Float16)(v[j] * stats[c] + stats[32 + c]);
    }
  }
  __syncthreads();
  for (int i = tid; i < 1024; i += 256)              // 8192 halfs contiguous
    *(v8h*)(feat + (size_t)b0 * 512 + 8 * i) = *(const v8h*)(((_Float16*)sF) + 8 * i);
}

// routing: per-sample argmax of 16 logits for all 3 stages, compact lists
__global__ __launch_bounds__(256) void k_route(
    const int* __restrict__ tasks, const float* __restrict__ p1,
    const float* __restrict__ p2, const float* __restrict__ p3,
    int* __restrict__ counts, int* __restrict__ lists)
{
  int b = blockIdx.x * 256 + threadIdx.x;
  if (b >= NB) return;
  int t = tasks[b];
  const float* ps[3] = { p1 + t * 16, p2 + t * 16, p3 + t * 16 };
#pragma unroll
  for (int s = 0; s < 3; ++s) {
    const float* p = ps[s];
    int e = 0; float best = p[0];
    for (int j = 1; j < 16; ++j) { float v = p[j]; if (v > best) { best = v; e = j; } }
    int pos = atomicAdd(&counts[s * 16 + e], 1);
    lists[(s * 16 + e) * NB + pos] = b;
  }
}

// ---------------------------------------------------------------------------
// fc1: 512->64 per-expert gathered GEMM, K chunked by 128, all staging via DMA.
// ---------------------------------------------------------------------------
__global__ __launch_bounds__(256) void k_fc1(
    const _Float16* __restrict__ feat, const _Float16* __restrict__ fw1,
    const float* __restrict__ b1, const int* __restrict__ counts,
    const int* __restrict__ lists, _Float16* __restrict__ out)
{
  const int ex = blockIdx.y;
  const int cnt = counts[ex];
  const int row0 = blockIdx.x * 64;
  if (row0 >= cnt) return;
  const int tid = threadIdx.x;
  __shared__ alignas(16) _Float16 sX[64][128];
  __shared__ alignas(16) _Float16 sW[64][128];
  __shared__ int sIdx[64];
  if (tid < 64) {
    int rr = row0 + tid;
    sIdx[tid] = lists[ex * NB + (rr < cnt ? rr : cnt - 1)];
  }
  __syncthreads();
  const int wave = tid >> 5, lane = tid & 31, sel = lane >> 4;
  v8f acc0, acc1;
  {
    int oc0 = ((wave)     & 3) * 16 + (lane & 15);
    int oc1 = ((wave + 8) & 3) * 16 + (lane & 15);
    float bb0 = b1[ex * 64 + oc0], bb1 = b1[ex * 64 + oc1];
#pragma unroll
    for (int i = 0; i < 8; ++i) { acc0[i] = bb0; acc1[i] = bb1; }
  }
  for (int k0 = 0; k0 < 512; k0 += 128) {
    for (int i = tid; i < 1024; i += 256) {          // W rows: 16B chunks
      int oc = i >> 4, q = i & 15;
      cp16(&sW[oc][8 * q], fw1 + (size_t)(ex * 64 + oc) * 512 + k0 + 8 * q);
    }
    for (int i = tid; i < 1024; i += 256) {          // X rows: 16B chunks
      int rr = i >> 4, q = i & 15;
      cp16(&sX[rr][8 * q], feat + (size_t)sIdx[rr] * 512 + k0 + 8 * q);
    }
    ASYNC_WAIT();
    __syncthreads();
#pragma unroll
    for (int ci = 0; ci < 2; ++ci) {
      int combo = wave + 8 * ci;
      int rowtile = combo >> 2, octile = combo & 3;
      v8f acc = ci ? acc1 : acc0;
#pragma unroll
      for (int ks = 0; ks < 4; ++ks) {
        int kb = ks * 32;
        int row = rowtile * 16 + (lane & 15);
        v16h a  = make_frag(*(const v8h*)&sX[row][kb + 16 * sel],
                            *(const v8h*)&sX[row][kb + 16 * sel + 8]);
        int oc = octile * 16 + (lane & 15);
        v16h bf = make_frag(*(const v8h*)&sW[oc][kb + 16 * sel],
                            *(const v8h*)&sW[oc][kb + 16 * sel + 8]);
        acc = __builtin_amdgcn_wmma_f32_16x16x32_f16(false, a, false, bf,
                                                     (short)0, acc, false, false);
      }
      if (ci) acc1 = acc; else acc0 = acc;
    }
    __syncthreads();
  }
#pragma unroll
  for (int ci = 0; ci < 2; ++ci) {
    int combo = wave + 8 * ci;
    int rowtile = combo >> 2, octile = combo & 3;
    v8f acc = ci ? acc1 : acc0;
    int oc = octile * 16 + (lane & 15);
#pragma unroll
    for (int i = 0; i < 8; ++i) {
      int row = rowtile * 16 + 8 * sel + i;
      if (row0 + row < cnt) out[sIdx[row] * 64 + oc] = (_Float16)acc[i];
    }
  }
}

// fc2: 64->64 per-expert gathered GEMM (W block fully contiguous)
__global__ __launch_bounds__(256) void k_fc2(
    const _Float16* __restrict__ in, const _Float16* __restrict__ fw2,
    const float* __restrict__ b2, const int* __restrict__ counts,
    const int* __restrict__ lists, _Float16* __restrict__ out)
{
  const int ex = blockIdx.y;
  const int cnt = counts[16 + ex];
  const int row0 = blockIdx.x * 64;
  if (row0 >= cnt) return;
  const int tid = threadIdx.x;
  __shared__ alignas(16) _Float16 sX[64][64];
  __shared__ alignas(16) _Float16 sW[64][64];
  __shared__ int sIdx[64];
  if (tid < 64) {
    int rr = row0 + tid;
    sIdx[tid] = lists[(16 + ex) * NB + (rr < cnt ? rr : cnt - 1)];
  }
  __syncthreads();
  for (int i = tid; i < 512; i += 256)               // 4096 halfs contiguous
    cp16(((_Float16*)sW) + 8 * i, fw2 + (size_t)ex * 4096 + 8 * i);
  for (int i = tid; i < 512; i += 256) {
    int rr = i >> 3, q = i & 7;
    cp16(&sX[rr][8 * q], in + (size_t)sIdx[rr] * 64 + 8 * q);
  }
  ASYNC_WAIT();
  __syncthreads();
  const int wave = tid >> 5, lane = tid & 31, sel = lane >> 4;
#pragma unroll
  for (int ci = 0; ci < 2; ++ci) {
    int combo = wave + 8 * ci;
    int rowtile = combo >> 2, octile = combo & 3;
    int oc = octile * 16 + (lane & 15);
    v8f acc;
    float bb = b2[ex * 64 + oc];
#pragma unroll
    for (int i = 0; i < 8; ++i) acc[i] = bb;
#pragma unroll
    for (int ks = 0; ks < 2; ++ks) {
      int kb = ks * 32;
      int row = rowtile * 16 + (lane & 15);
      v16h a  = make_frag(*(const v8h*)&sX[row][kb + 16 * sel],
                          *(const v8h*)&sX[row][kb + 16 * sel + 8]);
      v16h bf = make_frag(*(const v8h*)&sW[oc][kb + 16 * sel],
                          *(const v8h*)&sW[oc][kb + 16 * sel + 8]);
      acc = __builtin_amdgcn_wmma_f32_16x16x32_f16(false, a, false, bf,
                                                   (short)0, acc, false, false);
    }
#pragma unroll
    for (int i = 0; i < 8; ++i) {
      int row = rowtile * 16 + 8 * sel + i;
      if (row0 + row < cnt) out[sIdx[row] * 64 + oc] = (_Float16)acc[i];
    }
  }
}

// fc3: 64->10 (oc padded to 16 in fw3), writes final f32 output with bias
__global__ __launch_bounds__(256) void k_fc3(
    const _Float16* __restrict__ in, const _Float16* __restrict__ fw3,
    const float* __restrict__ b3, const int* __restrict__ counts,
    const int* __restrict__ lists, float* __restrict__ out)
{
  const int ex = blockIdx.y;
  const int cnt = counts[32 + ex];
  const int row0 = blockIdx.x * 64;
  if (row0 >= cnt) return;
  const int tid = threadIdx.x;
  __shared__ alignas(16) _Float16 sX[64][64];
  __shared__ alignas(16) _Float16 sW[16][64];
  __shared__ int sIdx[64];
  if (tid < 64) {
    int rr = row0 + tid;
    sIdx[tid] = lists[(32 + ex) * NB + (rr < cnt ? rr : cnt - 1)];
  }
  __syncthreads();
  for (int i = tid; i < 128; i += 256)               // 1024 halfs contiguous
    cp16(((_Float16*)sW) + 8 * i, fw3 + (size_t)ex * 1024 + 8 * i);
  for (int i = tid; i < 512; i += 256) {
    int rr = i >> 3, q = i & 7;
    cp16(&sX[rr][8 * q], in + (size_t)sIdx[rr] * 64 + 8 * q);
  }
  ASYNC_WAIT();
  __syncthreads();
  const int wave = tid >> 5, lane = tid & 31, sel = lane >> 4;
  if (wave < 4) {
    int rowtile = wave;
    v8f acc = {};
#pragma unroll
    for (int ks = 0; ks < 2; ++ks) {
      int kb = ks * 32;
      int row = rowtile * 16 + (lane & 15);
      v16h a  = make_frag(*(const v8h*)&sX[row][kb + 16 * sel],
                          *(const v8h*)&sX[row][kb + 16 * sel + 8]);
      int oc = lane & 15;
      v16h bf = make_frag(*(const v8h*)&sW[oc][kb + 16 * sel],
                          *(const v8h*)&sW[oc][kb + 16 * sel + 8]);
      acc = __builtin_amdgcn_wmma_f32_16x16x32_f16(false, a, false, bf,
                                                   (short)0, acc, false, false);
    }
    int oc = lane & 15;
    if (oc < 10) {
      float bb = b3[ex * 10 + oc];
#pragma unroll
      for (int i = 0; i < 8; ++i) {
        int row = rowtile * 16 + 8 * sel + i;
        if (row0 + row < cnt) out[sIdx[row] * 10 + oc] = acc[i] + bb;
      }
    }
  }
}

// ---------------------------------------------------------------------------
extern "C" void kernel_launch(void* const* d_in, const int* in_sizes, int n_in,
                              void* d_out, int out_size, void* d_ws, size_t ws_size,
                              hipStream_t stream) {
  const float* x     = (const float*)d_in[0];
  const int*   tasks = (const int*)  d_in[1];
  const float* Wc1   = (const float*)d_in[2];
  const float* bc1   = (const float*)d_in[3];
  const float* Wc2   = (const float*)d_in[4];
  const float* bc2   = (const float*)d_in[5];
  const float* Wc3   = (const float*)d_in[6];
  const float* bc3   = (const float*)d_in[7];
  const float* gamma = (const float*)d_in[8];
  const float* beta  = (const float*)d_in[9];
  const float* p1    = (const float*)d_in[10];
  const float* p2    = (const float*)d_in[11];
  const float* p3    = (const float*)d_in[12];
  const float* W1    = (const float*)d_in[13];
  const float* b1    = (const float*)d_in[14];
  const float* W2    = (const float*)d_in[15];
  const float* b2    = (const float*)d_in[16];
  const float* W3    = (const float*)d_in[17];
  const float* b3    = (const float*)d_in[18];

  char* ws = (char*)d_ws;
  _Float16* act1  = (_Float16*)(ws);                    // [B][16][16][32] f16
  _Float16* act2  = (_Float16*)(ws + 134217728ull);     // [B][8][8][32]  f16
  float*    act3  = (float*)   (ws + 167772160ull);     // [B][4][4][32]  f32
  _Float16* feat  = (_Float16*)(ws + 184549376ull);     // [B][512]       f16
  _Float16* y1    = (_Float16*)(ws + 192937984ull);     // [B][64]        f16
  _Float16* y2    = (_Float16*)(ws + 193986560ull);     // [B][64]        f16
  float*    stats = (float*)   (ws + 195035136ull);     // scale[32], shift[32]
  int*      counts= (int*)     (ws + 195035392ull);     // [3][16]
  int*      lists = (int*)     (ws + 195035584ull);     // [3][16][B]
  _Float16* fw1   = (_Float16*)(ws + 196608448ull);     // [16][64][512] f16
  _Float16* fw2   = (_Float16*)(ws + 197657024ull);     // [16][64][64]  f16
  _Float16* fw3   = (_Float16*)(ws + 197788096ull);     // [16][16][64]  f16
  _Float16* cw1   = (_Float16*)(ws + 197820864ull);     // [32][32]      f16
  _Float16* cw2   = (_Float16*)(ws + 197822912ull);     // [32][288]     f16
  _Float16* cw3   = (_Float16*)(ws + 197841344ull);     // [32][288]     f16
  float*    part  = (float*)   (ws + 197859776ull);     // [64][64] BN partials

  (void)hipMemsetAsync(counts, 0, 48 * sizeof(int), stream);

  k_prep<<<2444, 256, 0, stream>>>(W1, W2, W3, Wc1, Wc2, Wc3,
                                   fw1, fw2, fw3, cw1, cw2, cw3);
  k_conv1<<<NB * 2, 256, 0, stream>>>(x, cw1, bc1, act1);
  k_conv2<<<NB, 256, 0, stream>>>(act1, cw2, bc2, act2);
  k_conv3<<<NB, 256, 0, stream>>>(act2, cw3, bc3, act3);
  k_bnpart<<<64, 256, 0, stream>>>(act3, part);
  k_bnfinal<<<1, 32, 0, stream>>>(part, gamma, beta, stats);
  k_feat<<<NB / 16, 256, 0, stream>>>(act3, stats, feat);
  k_route<<<NB / 256, 256, 0, stream>>>(tasks, p1, p2, p3, counts, lists);
  dim3 g(NB / 64, 16);
  k_fc1<<<g, 256, 0, stream>>>(feat, fw1, b1, counts, lists, y1);
  k_fc2<<<g, 256, 0, stream>>>(y1, fw2, b2, counts, lists, y2);
  k_fc3<<<g, 256, 0, stream>>>(y2, fw3, b3, counts, lists, (float*)d_out);
}
